// SlayerMLP_28587302322492
// MI455X (gfx1250) — compile-verified, matrix operand order checked by reading, not run on hardware
//
#include <hip/hip_runtime.h>

// SLAYER 2-layer SNN forward, CDNA5 (gfx1250) WMMA implementation, v2.
// - Layer matmuls AND the K=100 causal SRM convolutions run on the matrix
//   pipe (v_wmma_f32_16x16x32_f16); the conv is 4 chained WMMAs against
//   constant Toeplitz blocks of the alpha kernel.
// - All global reads are coalesced + branch-free (clamped address + select,
//   never exec-predicated loads); operands are staged into LDS in
//   fragment-ready layout so每 fragment is built with two ds_load_b128.

typedef __attribute__((ext_vector_type(16))) _Float16 v16h;
typedef __attribute__((ext_vector_type(8)))  _Float16 v8h;
typedef __attribute__((ext_vector_type(2)))  _Float16 v2h;
typedef __attribute__((ext_vector_type(8)))  float    v8f;

#define B_     256
#define T_     1000
#define IN_    78
#define HID_   16
#define OUT_   20
#define KS_    100      // SRM kernel support
#define TILE_T 128      // output timesteps per workgroup (8 waves x 16 cols)
#define HALO   112      // conv back-halo: 112+16 = 128 = 4 K-chunks of 32
#define ZCOLS  (TILE_T + HALO)   // 240 pre-conv columns staged in LDS
#define GTILES (ZCOLS / 16)      // 15 sixteen-wide column tiles

__device__ __forceinline__ int iclamp(int v, int lo, int hi) {
    return v < lo ? lo : (v > hi ? hi : v);
}

// SLAYER SRMALPHA kernel: eps(s) = (s/tau) * exp(1 - s/tau), tau = 10
__device__ __forceinline__ float srm_eps(int s) {
    if (s < 0 || s >= KS_) return 0.0f;
    float x = (float)s * 0.1f;
    return x * __expf(1.0f - x);
}

__device__ __forceinline__ v8f wmma_f16(v16h a, v16h b, v8f c) {
    return __builtin_amdgcn_wmma_f32_16x16x32_f16(
        /*neg_a=*/false, a, /*neg_b=*/false, b,
        /*c_mod=*/(short)0, c, /*reuse_a=*/false, /*reuse_b=*/false);
}

// Per the CDNA5 16-bit A/B fragment layout (CDNA5 ISA §7.12.2) a
// lane's 16 half-slots cover K = {loff..loff+7, loff+16..loff+23} within a
// 32-wide K chunk (loff = 8 for lanes 16-31). With operands staged K-contiguous
// in LDS, a fragment is exactly two 16-byte ds loads.
__device__ __forceinline__ v16h frag16(const _Float16* p) {
    v8h lo = *(const v8h*)(p);
    v8h hi = *(const v8h*)(p + 16);
    return __builtin_shufflevector(lo, hi, 0, 1, 2, 3, 4, 5, 6, 7,
                                          8, 9, 10, 11, 12, 13, 14, 15);
}

// K index within a 32-wide K-chunk for half-slot h of this lane.
__device__ __forceinline__ int koff(int h, int lane) {
    return h + ((h >= 8) ? 8 : 0) + ((lane >= 16) ? 8 : 0);
}

// The 4 constant Toeplitz B-fragments of the SRM kernel for this lane.
__device__ __forceinline__ void build_eps_frags(v16h e[4], int lane) {
    const int j = lane & 15;  // N (output time within 16-tile)
    for (int c = 0; c < 4; ++c)
        for (int h = 0; h < 16; ++h)
            e[c][h] = (_Float16)srm_eps(HALO - 32 * c + j - koff(h, lane));
}

// ---------------- layer 1: spikes -> s1 (f16 in workspace) ----------------
__global__ __launch_bounds__(256) void slayer_layer1(
    const float* __restrict__ x,      // [B, IN, T] f32 (binary spikes)
    const float* __restrict__ W1,     // [HID, IN]  f32
    _Float16*    __restrict__ s1)     // [B, HID, T] f16 (output)
{
    __shared__ __align__(16) _Float16 xsh[ZCOLS * 96];  // x tile, [col][chan] (45 KB)
    __shared__ __align__(16) _Float16 wsh[16 * 96];     // W1 f16, IN padded ->96
    __shared__ __align__(16) _Float16 zsh[16 * ZCOLS];  // pre-conv z1

    const int lane = threadIdx.x & 31;
    const int wave = threadIdx.x >> 5;
    const int b    = blockIdx.y;
    const int t0   = blockIdx.x * TILE_T;
    const int ln   = lane & 15;
    const int loff = (lane >= 16) ? 8 : 0;
    const float* xb = x + (size_t)b * IN_ * T_;

    // Warm L2 with this block's input stripe (global_prefetch_b8 path).
    {
        int tp = iclamp(t0 - HALO, 0, T_ - 1);
        for (int i = threadIdx.x; i < IN_; i += 256)
            __builtin_prefetch(&xb[(size_t)i * T_ + tp], 0, 1);
    }

    v16h e[4];
    build_eps_frags(e, lane);

    // Stage W1 (f16, padded) — clamped address + select, branch-free.
    for (int idx = threadIdx.x; idx < 16 * 96; idx += 256) {
        const int h = idx / 96, i = idx % 96;
        const float w = W1[h * IN_ + iclamp(i, 0, IN_ - 1)];
        wsh[idx] = (i < IN_) ? (_Float16)w : (_Float16)0.0f;
    }

    // Stage x tile transposed: xsh[col*96 + chan]. Consecutive threads take
    // consecutive cols (coalesced along t); channels paired for packed stores.
    for (int idx = threadIdx.x; idx < 48 * ZCOLS; idx += 256) {
        const int col = idx % ZCOLS;
        const int i   = (idx / ZCOLS) * 2;
        const int t   = t0 - HALO + col;
        const int tc  = iclamp(t, 0, T_ - 1);
        const bool tv = (t >= 0) & (t < T_);
        const float v0 = xb[(size_t)i * T_ + tc];
        const float v1 = xb[(size_t)iclamp(i + 1, 0, IN_ - 1) * T_ + tc];
        v2h p;
        p[0] = (tv && i     < IN_) ? (_Float16)v0 : (_Float16)0.0f;
        p[1] = (tv && i + 1 < IN_) ? (_Float16)v1 : (_Float16)0.0f;
        *(v2h*)&xsh[col * 96 + i] = p;
    }
    __syncthreads();

    // z1 = W1 @ x into LDS: 15 column tiles, K = 96 (3 chunks of 32).
    for (int g = wave; g < GTILES; g += 8) {            // wave-uniform
        v8f acc = {};
        const int col = g * 16 + ln;                    // N = time column
        for (int kc = 0; kc < 3; ++kc) {
            v16h a  = frag16(&wsh[ln * 96 + kc * 32 + loff]);   // M=hid, K=chan
            v16h bb = frag16(&xsh[col * 96 + kc * 32 + loff]);  // K=chan, N=t
            acc = wmma_f16(a, bb, acc);
        }
        for (int r = 0; r < 8; ++r)
            zsh[(r + loff) * ZCOLS + col] = (_Float16)acc[r];
    }
    __syncthreads();

    // Toeplitz-WMMA conv + threshold -> s1 for this wave's 16 output columns.
    v8f u = {};
    for (int c = 0; c < 4; ++c) {
        v16h a = frag16(&zsh[ln * ZCOLS + wave * 16 + 32 * c + loff]);
        u = wmma_f16(a, e[c], u);
    }
    const int t = t0 + wave * 16 + ln;
    if (t < T_) {
        for (int r = 0; r < 8; ++r)
            s1[(size_t)b * HID_ * T_ + (size_t)(r + loff) * T_ + t] =
                (u[r] >= 1.0f) ? (_Float16)1.0f : (_Float16)0.0f;
    }
}

// ---------------- layer 2: s1 -> output spikes (f32) ----------------
__global__ __launch_bounds__(256) void slayer_layer2(
    const _Float16* __restrict__ s1,  // [B, HID, T] f16
    const float*    __restrict__ W2,  // [OUT, HID]  f32
    float*          __restrict__ out) // [B, OUT, T] f32
{
    __shared__ __align__(16) _Float16 s1sh[ZCOLS * 32]; // s1 tile, [col][hid pad 32]
    __shared__ __align__(16) _Float16 wsh2[32 * 32];    // W2 padded 20->32, 16->32
    __shared__ __align__(16) _Float16 zsh[32 * ZCOLS];  // pre-conv z2

    const int lane = threadIdx.x & 31;
    const int wave = threadIdx.x >> 5;
    const int b    = blockIdx.y;
    const int t0   = blockIdx.x * TILE_T;
    const int ln   = lane & 15;
    const int loff = (lane >= 16) ? 8 : 0;
    const _Float16* s1b = s1 + (size_t)b * HID_ * T_;

    v16h e[4];
    build_eps_frags(e, lane);

    // Stage W2 padded (branch-free).
    for (int idx = threadIdx.x; idx < 32 * 32; idx += 256) {
        const int row = idx / 32, k = idx % 32;
        const float w = W2[iclamp(row, 0, OUT_ - 1) * HID_ + iclamp(k, 0, HID_ - 1)];
        wsh2[idx] = (row < OUT_ && k < HID_) ? (_Float16)w : (_Float16)0.0f;
    }
    // Stage s1 tile transposed: s1sh[col*32 + hid]; pad rows 16..31 with zero.
    for (int idx = threadIdx.x; idx < 16 * ZCOLS; idx += 256) {
        const int col = idx % ZCOLS;
        const int h   = idx / ZCOLS;
        const int t   = t0 - HALO + col;
        const _Float16 v = s1b[(size_t)h * T_ + iclamp(t, 0, T_ - 1)];
        s1sh[col * 32 + h]      = (t >= 0 && t < T_) ? v : (_Float16)0.0f;
        s1sh[col * 32 + 16 + h] = (_Float16)0.0f;
    }
    __syncthreads();

    // z2 = W2 @ s1 into LDS: 2 M-tiles x 15 column tiles, K = 32 (padded).
    for (int job = wave; job < 2 * GTILES; job += 8) {  // wave-uniform
        const int m   = job / GTILES;
        const int g   = job % GTILES;
        const int col = g * 16 + ln;
        v16h a  = frag16(&wsh2[(m * 16 + ln) * 32 + loff]);  // M=out, K=hid
        v16h bb = frag16(&s1sh[col * 32 + loff]);            // K=hid, N=t
        v8f acc = {};
        acc = wmma_f16(a, bb, acc);
        for (int r = 0; r < 8; ++r)
            zsh[(m * 16 + r + loff) * ZCOLS + col] = (_Float16)acc[r];
    }
    __syncthreads();

    // Conv + threshold, two 16-row M-tiles covering the 20 output channels.
    const int t = t0 + wave * 16 + ln;
    for (int m = 0; m < 2; ++m) {
        v8f u = {};
        for (int c = 0; c < 4; ++c) {
            v16h a = frag16(&zsh[(m * 16 + ln) * ZCOLS + wave * 16 + 32 * c + loff]);
            u = wmma_f16(a, e[c], u);
        }
        if (t < T_) {
            for (int r = 0; r < 8; ++r) {
                const int o = m * 16 + r + loff;
                if (o < OUT_)
                    out[(size_t)b * OUT_ * T_ + (size_t)o * T_ + t] =
                        (u[r] >= 1.0f) ? 1.0f : 0.0f;
            }
        }
    }
}

extern "C" void kernel_launch(void* const* d_in, const int* in_sizes, int n_in,
                              void* d_out, int out_size, void* d_ws, size_t ws_size,
                              hipStream_t stream) {
    const float* x  = (const float*)d_in[0];   // [256, 78, 1000]
    const float* W1 = (const float*)d_in[1];   // [16, 78]
    const float* W2 = (const float*)d_in[2];   // [20, 16]
    float*       out = (float*)d_out;          // [256, 20, 1000]
    _Float16*    s1  = (_Float16*)d_ws;        // [256, 16, 1000] f16 = 8.2 MB

    dim3 grid((T_ + TILE_T - 1) / TILE_T, B_);  // 8 x 256 workgroups
    slayer_layer1<<<grid, 256, 0, stream>>>(x, W1, s1);
    slayer_layer2<<<grid, 256, 0, stream>>>(s1, W2, out);
}
